// GATEo9bn_55748675502669
// MI455X (gfx1250) — matrix operand order repaired and checked by reading, not compiled, not run on hardware
//
#include <hip/hip_runtime.h>
#include <hip/hip_bf16.h>

// ---------------------------------------------------------------------------
// Problem constants (match reference)
// ---------------------------------------------------------------------------
#define NN 40000    // nodes
#define NE 160000   // edges
#define NG 128      // graphs
#define BN_EPS 1e-5f

typedef __attribute__((ext_vector_type(16))) __bf16 v16bf;
typedef __attribute__((ext_vector_type(8)))  float  v8f;

struct U128 { unsigned int x, y, z, w; };   // POD 16-byte chunk (union-safe)

static __device__ __forceinline__ __bf16 f2bf(float f) {
    union { float f; unsigned u; } c; c.f = f;
    unsigned r = c.u + 0x7FFFu + ((c.u >> 16) & 1u);   // round-to-nearest-even
    unsigned short h = (unsigned short)(r >> 16);
    return __builtin_bit_cast(__bf16, h);
}

// LDS byte offset of a __shared__ element (generic LDS ptr low 32 bits = LDS addr)
static __device__ __forceinline__ unsigned lds_off(const void* p) {
    return (unsigned)(unsigned long long)p;
}

// ---------------------------------------------------------------------------
// Core WMMA GEMM: C[M,N] = act(A[M,Kp](bf16) @ W[Kp,N] + bias (+resid))
// Weights pre-transposed: Wt[N][Kp] bf16.  Tile 128x64, K-step 32, 8 waves.
// Full tiles: double-buffered LDS fed by global_load_async_to_lds_b128.
// ---------------------------------------------------------------------------
#define TM 128
#define TN 64
#define TK 32
#define APAD 8     // bf16 pad -> row stride 40 (80B), conflict-free 16B chunks

__global__ __launch_bounds__(256) void wmma_gemm(
    const __bf16* __restrict__ A, const __bf16* __restrict__ Wt,
    const float* __restrict__ bias, const float* __restrict__ resid,
    float* __restrict__ C, int M, int N, int Kp, int relu)
{
    __shared__ __bf16 sA[2][TM][TK + APAD];
    __shared__ __bf16 sB[2][TN][TK + APAD];

    const int tid  = threadIdx.x;
    const int wave = tid >> 5;
    const int lane = tid & 31;
    const int half = lane >> 4;     // 0: lanes 0-15, 1: lanes 16-31
    const int l15  = lane & 15;
    const long m0 = (long)blockIdx.x * TM;
    const int  n0 = blockIdx.y * TN;

    v8f acc[4] = {};
    const int ktiles = Kp / TK;

    // per-thread fixed chunk coordinates
    const int ar0 = tid >> 2;          // A chunk 1 row (0..63)
    const int ar1 = (tid >> 2) + 64;   // A chunk 2 row (64..127)
    const int acc4 = (tid & 3) * 8;    // bf16 column offset of 16B chunk
    const int br  = tid >> 2;          // B row (0..63)

    if (m0 + TM <= M) {
        // ---------------- full-tile fast path: async LDS staging ------------
        const __bf16* Abase = A + m0 * (long)Kp + acc4;
        const __bf16* Bbase = Wt + (long)(n0 + br) * Kp + acc4;

        auto stage = [&](int kt, int buf) {
            unsigned la0 = lds_off(&sA[buf][ar0][acc4]);
            unsigned long long ga0 =
                (unsigned long long)(Abase + (long)ar0 * Kp + kt * TK);
            asm volatile("global_load_async_to_lds_b128 %0, %1, off"
                         :: "v"(la0), "v"(ga0) : "memory");
            unsigned la1 = lds_off(&sA[buf][ar1][acc4]);
            unsigned long long ga1 =
                (unsigned long long)(Abase + (long)ar1 * Kp + kt * TK);
            asm volatile("global_load_async_to_lds_b128 %0, %1, off"
                         :: "v"(la1), "v"(ga1) : "memory");
            unsigned lb = lds_off(&sB[buf][br][acc4]);
            unsigned long long gb =
                (unsigned long long)(Bbase + kt * TK);
            asm volatile("global_load_async_to_lds_b128 %0, %1, off"
                         :: "v"(lb), "v"(gb) : "memory");
        };

        stage(0, 0);
        for (int kt = 0; kt < ktiles; ++kt) {
            const int buf = kt & 1;
            if (kt + 1 < ktiles) {
                stage(kt + 1, buf ^ 1);
                asm volatile("s_wait_asynccnt 0x3" ::: "memory"); // tile kt done
            } else {
                asm volatile("s_wait_asynccnt 0x0" ::: "memory");
            }
            __syncthreads();

            // A fragment (ISA 16-bit 16x32): lanes<16 K0..7|16..23, lanes>=16 K8..15|24..31
            union { v16bf v; U128 q[2]; } af, bfj[4];
            af.q[0] = *(const U128*)(&sA[buf][wave * 16 + l15][half * 8]);
            af.q[1] = *(const U128*)(&sA[buf][wave * 16 + l15][half * 8 + 16]);
            // B fragments (32x16): lanes<16 K0..7|16..23, lanes>=16 K8..15|24..31
            #pragma unroll
            for (int j = 0; j < 4; ++j) {
                bfj[j].q[0] = *(const U128*)(&sB[buf][j * 16 + l15][half * 8]);
                bfj[j].q[1] = *(const U128*)(&sB[buf][j * 16 + l15][half * 8 + 16]);
            }
            #pragma unroll
            for (int j = 0; j < 4; ++j)
                acc[j] = __builtin_amdgcn_wmma_f32_16x16x32_bf16(
                    false, af.v, false, bfj[j].v, (short)0, acc[j], false, false);
            __syncthreads();
        }
    } else {
        // ---------------- partial-tile path: guarded synchronous staging ----
        for (int kt = 0; kt < ktiles; ++kt) {
            const int k0 = kt * TK;
            {
                U128 v0 = {0u,0u,0u,0u}, v1 = {0u,0u,0u,0u};
                if (m0 + ar0 < M)
                    v0 = *(const U128*)(A + (m0 + ar0) * (long)Kp + k0 + acc4);
                if (m0 + ar1 < M)
                    v1 = *(const U128*)(A + (m0 + ar1) * (long)Kp + k0 + acc4);
                *(U128*)(&sA[0][ar0][acc4]) = v0;
                *(U128*)(&sA[0][ar1][acc4]) = v1;
                *(U128*)(&sB[0][br][acc4]) =
                    *(const U128*)(Wt + (long)(n0 + br) * Kp + k0 + acc4);
            }
            __syncthreads();
            union { v16bf v; U128 q[2]; } af, bfj[4];
            af.q[0] = *(const U128*)(&sA[0][wave * 16 + l15][half * 8]);
            af.q[1] = *(const U128*)(&sA[0][wave * 16 + l15][half * 8 + 16]);
            #pragma unroll
            for (int j = 0; j < 4; ++j) {
                bfj[j].q[0] = *(const U128*)(&sB[0][j * 16 + l15][half * 8]);
                bfj[j].q[1] = *(const U128*)(&sB[0][j * 16 + l15][half * 8 + 16]);
            }
            #pragma unroll
            for (int j = 0; j < 4; ++j)
                acc[j] = __builtin_amdgcn_wmma_f32_16x16x32_bf16(
                    false, af.v, false, bfj[j].v, (short)0, acc[j], false, false);
            __syncthreads();
        }
    }

    // ---- store: C/D layout: VGPR r -> row (half*8 + r), col = lane&15
    #pragma unroll
    for (int j = 0; j < 4; ++j) {
        int gn = n0 + j * 16 + l15;
        float bv = bias ? bias[gn] : 0.f;
        #pragma unroll
        for (int r = 0; r < 8; ++r) {
            long gm = m0 + wave * 16 + half * 8 + r;
            if (gm < M) {
                float v = acc[j][r] + bv;
                if (resid) v += resid[gm * (long)N + gn];
                if (relu)  v = v > 0.f ? v : 0.f;
                C[gm * (long)N + gn] = v;
            }
        }
    }
}

// ---------------------------------------------------------------------------
// A-matrix builder: bf16 A[M][Kp] = concat(S0[I0[m]|m], S1[I1[m]|m], S2[m]), 0-pad
// ---------------------------------------------------------------------------
__global__ void build3(const float* __restrict__ S0, const int* __restrict__ I0, int f0,
                       const float* __restrict__ S1, const int* __restrict__ I1, int f1,
                       const float* __restrict__ S2, int f2,
                       __bf16* __restrict__ A, long M, int Kp)
{
    long total = M * Kp;
    long stride = (long)gridDim.x * blockDim.x;
    for (long i = (long)blockIdx.x * blockDim.x + threadIdx.x; i < total; i += stride) {
        long m = i / Kp;
        int  k = (int)(i - m * Kp);
        float v = 0.f;
        if (k < f0) {
            long r = I0 ? (long)I0[m] : m;
            v = S0[r * f0 + k];
        } else if (k < f0 + f1) {
            long r = I1 ? (long)I1[m] : m;
            v = S1[r * f1 + (k - f0)];
        } else if (k < f0 + f1 + f2) {
            v = S2[m * (long)f2 + (k - f0 - f1)];
        }
        A[i] = f2bf(v);
    }
}

// Weight transpose+pad: W[K][N] f32 -> Wt[N][Kp] bf16
__global__ void wt_prep(const float* __restrict__ W, __bf16* __restrict__ Wt,
                        int K, int N, int Kp)
{
    long total = (long)N * Kp;
    long stride = (long)gridDim.x * blockDim.x;
    for (long i = (long)blockIdx.x * blockDim.x + threadIdx.x; i < total; i += stride) {
        int n = (int)(i / Kp), k = (int)(i - (long)n * Kp);
        Wt[i] = (k < K) ? f2bf(W[(long)k * N + n]) : f2bf(0.f);
    }
}

__global__ void zero_f32(float* __restrict__ p, long n) {
    long stride = (long)gridDim.x * blockDim.x;
    for (long i = (long)blockIdx.x * blockDim.x + threadIdx.x; i < n; i += stride)
        p[i] = 0.f;
}

// segment scatter-add (+count); s = seg2 ? seg2[seg[m]] : seg[m]
__global__ void seg_scatter(const float* __restrict__ V, const int* __restrict__ seg,
                            const int* __restrict__ seg2,
                            float* __restrict__ acc, float* __restrict__ cnt,
                            long M, int F)
{
    long total = M * F;
    long stride = (long)gridDim.x * blockDim.x;
    for (long i = (long)blockIdx.x * blockDim.x + threadIdx.x; i < total; i += stride) {
        long m = i / F;
        int  f = (int)(i - m * F);
        int  s = seg[m];
        if (seg2) s = seg2[s];
        atomicAdd(&acc[(long)s * F + f], V[i]);
        if (f == 0) atomicAdd(&cnt[s], 1.0f);
    }
}

__global__ void seg_div(float* __restrict__ acc, const float* __restrict__ cnt,
                        long S, int F)
{
    long total = S * F;
    long stride = (long)gridDim.x * blockDim.x;
    for (long i = (long)blockIdx.x * blockDim.x + threadIdx.x; i < total; i += stride) {
        long s = i / F;
        float c = cnt[s];
        acc[i] /= (c > 1.f ? c : 1.f);
    }
}

// BN: per-column sum & sumsq (block-partial + atomics into stats[0..F), stats[F..2F))
__global__ __launch_bounds__(256) void bn_stats(const float* __restrict__ H,
                                                long M, int F, float* __restrict__ stats)
{
    int chunk = (int)((M + gridDim.x - 1) / gridDim.x);
    long r0 = (long)blockIdx.x * chunk;
    long r1 = r0 + chunk; if (r1 > M) r1 = M;
    int c0 = threadIdx.x;
    float sm0 = 0.f, sq0 = 0.f, sm1 = 0.f, sq1 = 0.f;
    for (long r = r0; r < r1; ++r) {
        if (c0 < F)       { float v = H[r * F + c0];        sm0 += v; sq0 += v * v; }
        if (c0 + 256 < F) { float v = H[r * F + c0 + 256];  sm1 += v; sq1 += v * v; }
    }
    if (c0 < F)       { atomicAdd(&stats[c0], sm0);        atomicAdd(&stats[F + c0], sq0); }
    if (c0 + 256 < F) { atomicAdd(&stats[c0 + 256], sm1);  atomicAdd(&stats[F + c0 + 256], sq1); }
}

__global__ void bn_apply(float* __restrict__ H, long M, int F,
                         const float* __restrict__ stats,
                         const float* __restrict__ gamma, const float* __restrict__ beta)
{
    long total = M * F;
    long stride = (long)gridDim.x * blockDim.x;
    float invM = 1.0f / (float)M;
    for (long i = (long)blockIdx.x * blockDim.x + threadIdx.x; i < total; i += stride) {
        int c = (int)(i % F);
        float mean = stats[c] * invM;
        float var  = stats[F + c] * invM - mean * mean;
        float inv  = rsqrtf(var + BN_EPS);
        H[i] = gamma[c] * (H[i] - mean) * inv + beta[c];
    }
}

// Global-model MLP: Uout[g] = W2^T relu(W1^T concat(U[g], EG[g]) + b1) + b2
__global__ __launch_bounds__(128) void glob_mlp(
    const float* __restrict__ U, const float* __restrict__ EG, int gf, int eo,
    const float* __restrict__ W1, const float* __restrict__ b1, int h1,
    const float* __restrict__ W2, const float* __restrict__ b2, int gout,
    float* __restrict__ Uout)
{
    __shared__ float sin_[384];
    __shared__ float sh_[128];
    int g = blockIdx.x, t = threadIdx.x;
    int din = gf + eo;
    for (int k = t; k < din; k += 128)
        sin_[k] = (k < gf) ? U[(long)g * gf + k] : EG[(long)g * eo + (k - gf)];
    __syncthreads();
    if (t < h1) {
        float a = b1[t];
        for (int k = 0; k < din; ++k) a += sin_[k] * W1[(long)k * h1 + t];
        sh_[t] = a > 0.f ? a : 0.f;
    }
    __syncthreads();
    if (t < gout) {
        float a = b2[t];
        for (int k = 0; k < h1; ++k) a += sh_[k] * W2[(long)k * gout + t];
        Uout[(long)g * gout + t] = a;
    }
}

// Final head: out[g] = fc2( relu(fc1(u3[g])) )
__global__ __launch_bounds__(32) void fc_head(const float* __restrict__ U,
                                              const float* __restrict__ W1, const float* __restrict__ b1,
                                              const float* __restrict__ W2, const float* __restrict__ b2,
                                              float* __restrict__ out)
{
    __shared__ float h[16];
    int g = blockIdx.x, t = threadIdx.x;
    if (t < 16) {
        float a = b1[t];
        for (int k = 0; k < 64; ++k) a += U[(long)g * 64 + k] * W1[k * 16 + t];
        h[t] = a > 0.f ? a : 0.f;
    }
    __syncthreads();
    if (t == 0) {
        float a = b2[0];
        for (int k = 0; k < 16; ++k) a += h[k] * W2[k];
        out[g] = a;
    }
}

// ---------------------------------------------------------------------------
// Host-side orchestration
// ---------------------------------------------------------------------------
static inline int kp32(int k) { return (k + 31) & ~31; }
static inline unsigned gsz(long n) {
    long b = (n + 255) / 256;
    if (b > 1048576) b = 1048576;
    if (b < 1) b = 1;
    return (unsigned)b;
}

static void launch_gemm(const __bf16* A, const __bf16* Wt, const float* bias,
                        const float* resid, float* C, long M, int N, int Kp,
                        int relu, hipStream_t s)
{
    dim3 grid((unsigned)((M + TM - 1) / TM), (unsigned)(N / TN));
    wmma_gemm<<<grid, 256, 0, s>>>(A, Wt, bias, resid, C, (int)M, N, Kp, relu);
}

struct Spec { int nf, ef, nh, eh, no, eo, gf; };

extern "C" void kernel_launch(void* const* d_in, const int* in_sizes, int n_in,
                              void* d_out, int out_size, void* d_ws, size_t ws_size,
                              hipStream_t stream)
{
    (void)in_sizes; (void)n_in; (void)out_size; (void)ws_size;

    // ---- inputs (setup_inputs dict order; params pytree: sorted dict keys) ----
    const float* x         = (const float*)d_in[0];
    const float* edge_attr = (const float*)d_in[1];
    const int*   eidx      = (const int*)d_in[2];
    const int*   rowIdx    = eidx;          // edge_index[0]
    const int*   colIdx    = eidx + NE;     // edge_index[1]
    const int*   batch     = (const int*)d_in[3];
    // d_in[4] = num_graphs (constant NG)
    // params leaves (sorted keys): edge_bn1, edge_bn2, fc1, fc2, l0, l1, l2, node_bn1, node_bn2
    const float* ebn1_g = (const float*)d_in[5];
    const float* ebn1_b = (const float*)d_in[6];
    const float* ebn2_g = (const float*)d_in[7];
    const float* ebn2_b = (const float*)d_in[8];
    const float* fc1_w  = (const float*)d_in[9];
    const float* fc1_b  = (const float*)d_in[10];
    const float* fc2_w  = (const float*)d_in[11];
    const float* fc2_b  = (const float*)d_in[12];
    const int LBASE[3] = {13, 29, 45};   // l{i}: edge1,edge2,glob1,glob2,node11,node12,node21,node22 (w,b each)
    const float* nbn1_g = (const float*)d_in[61];
    const float* nbn1_b = (const float*)d_in[62];
    const float* nbn2_g = (const float*)d_in[63];
    const float* nbn2_b = (const float*)d_in[64];

    Spec specs[3] = {{64,16,128,64,256,128,64},
                     {256,128,256,128,512,256,64},
                     {512,256,512,256,512,256,64}};

    // ---- workspace carve ----
    char* ws = (char*)d_ws;
    size_t off = 0;
    auto carve = [&](size_t bytes) -> void* {
        void* p = ws + off;
        off += (bytes + 255) & ~(size_t)255;
        return p;
    };
    __bf16* ABUF = (__bf16*)carve((size_t)NE * 1280 * 2);
    float*  F1   = (float*) carve((size_t)NE * 512 * 4);
    float*  F2   = (float*) carve((size_t)NE * 512 * 4);
    float*  EB0  = (float*) carve((size_t)NE * 256 * 4);
    float*  EB1  = (float*) carve((size_t)NE * 256 * 4);
    float*  XB0  = (float*) carve((size_t)NN * 512 * 4);
    float*  XB1  = (float*) carve((size_t)NN * 512 * 4);
    float*  AGG  = (float*) carve((size_t)NN * 512 * 4);
    float*  CNT  = (float*) carve((size_t)NN * 4);
    float*  EG   = (float*) carve((size_t)NG * 256 * 4);
    float*  GCNT = (float*) carve((size_t)NG * 4);
    float*  U0   = (float*) carve((size_t)NG * 64 * 4);
    float*  U1   = (float*) carve((size_t)NG * 64 * 4);
    float*  STAT = (float*) carve((size_t)2 * 512 * 4);
    __bf16* WT   = (__bf16*)carve((size_t)8 * 1024 * 1024);

    // ---- transpose+pad all 18 GEMM weights to bf16 ----
    struct LinW { const float* w; const float* b; int k, n, kp; __bf16* wt; };
    LinW lw[3][6];
    {
        size_t woff = 0;
        for (int i = 0; i < 3; ++i) {
            Spec s = specs[i];
            int kk[6] = {2*s.nf + s.ef, s.eh, s.nf + s.eo, s.nh, s.nf + s.nh, s.nh};
            int nn[6] = {s.eh, s.eo, s.nh, s.nh, s.nh, s.no};
            int li[6] = {0, 2, 8, 10, 12, 14};   // edge1,edge2,node11,node12,node21,node22
            for (int j = 0; j < 6; ++j) {
                LinW& L = lw[i][j];
                L.w = (const float*)d_in[LBASE[i] + li[j]];
                L.b = (const float*)d_in[LBASE[i] + li[j] + 1];
                L.k = kk[j]; L.n = nn[j]; L.kp = kp32(kk[j]);
                L.wt = WT + woff;
                woff += (size_t)L.n * L.kp;
                long tot = (long)L.n * L.kp;
                wt_prep<<<gsz(tot), 256, 0, stream>>>(L.w, L.wt, L.k, L.n, L.kp);
            }
        }
    }

    // u = zeros
    zero_f32<<<gsz(NG * 64), 256, 0, stream>>>(U0, NG * 64);

    // ---- one meta layer ----
    auto run_layer = [&](int i, const float* Xin, const float* Ein, const float* Uin,
                         float* Xout, float* Eout, float* Uout, bool R) {
        Spec s = specs[i];
        const float* gw1 = (const float*)d_in[LBASE[i] + 4];
        const float* gb1 = (const float*)d_in[LBASE[i] + 5];
        const float* gw2 = (const float*)d_in[LBASE[i] + 6];
        const float* gb2 = (const float*)d_in[LBASE[i] + 7];
        int h1 = s.eo / 2;

        // EdgeModel: e = mlp(concat(x[row], x[col], edge_attr)) (+Ein)
        int kpE = kp32(2 * s.nf + s.ef);
        build3<<<gsz((long)NE * kpE), 256, 0, stream>>>(
            Xin, rowIdx, s.nf, Xin, colIdx, s.nf, Ein, s.ef, ABUF, NE, kpE);
        launch_gemm(ABUF, lw[i][0].wt, lw[i][0].b, nullptr, F1, NE, s.eh, kpE, 1, stream);
        int kpH = kp32(s.eh);
        build3<<<gsz((long)NE * kpH), 256, 0, stream>>>(
            F1, nullptr, s.eh, nullptr, nullptr, 0, nullptr, 0, ABUF, NE, kpH);
        launch_gemm(ABUF, lw[i][1].wt, lw[i][1].b, R ? Ein : nullptr, Eout, NE, s.eo, kpH, 0, stream);

        // NodeModel part 1: n1 = mlp(concat(x[col], e)); agg = seg_mean(n1, row)
        int kpN1 = kp32(s.nf + s.eo);
        build3<<<gsz((long)NE * kpN1), 256, 0, stream>>>(
            Xin, colIdx, s.nf, Eout, nullptr, s.eo, nullptr, 0, ABUF, NE, kpN1);
        launch_gemm(ABUF, lw[i][2].wt, lw[i][2].b, nullptr, F1, NE, s.nh, kpN1, 1, stream);
        int kpNh = kp32(s.nh);
        build3<<<gsz((long)NE * kpNh), 256, 0, stream>>>(
            F1, nullptr, s.nh, nullptr, nullptr, 0, nullptr, 0, ABUF, NE, kpNh);
        launch_gemm(ABUF, lw[i][3].wt, lw[i][3].b, nullptr, F2, NE, s.nh, kpNh, 0, stream);

        zero_f32<<<gsz((long)NN * s.nh), 256, 0, stream>>>(AGG, (long)NN * s.nh);
        zero_f32<<<gsz(NN), 256, 0, stream>>>(CNT, NN);
        seg_scatter<<<gsz((long)NE * s.nh), 256, 0, stream>>>(F2, rowIdx, nullptr, AGG, CNT, NE, s.nh);
        seg_div<<<gsz((long)NN * s.nh), 256, 0, stream>>>(AGG, CNT, NN, s.nh);

        // NodeModel part 2: xn = mlp(concat(x, agg)) (+Xin)
        int kpN2 = kp32(s.nf + s.nh);
        build3<<<gsz((long)NN * kpN2), 256, 0, stream>>>(
            Xin, nullptr, s.nf, AGG, nullptr, s.nh, nullptr, 0, ABUF, NN, kpN2);
        launch_gemm(ABUF, lw[i][4].wt, lw[i][4].b, nullptr, F1, NN, s.nh, kpN2, 1, stream);
        build3<<<gsz((long)NN * kpNh), 256, 0, stream>>>(
            F1, nullptr, s.nh, nullptr, nullptr, 0, nullptr, 0, ABUF, NN, kpNh);
        launch_gemm(ABUF, lw[i][5].wt, lw[i][5].b, R ? Xin : nullptr, Xout, NN, s.no, kpNh, 0, stream);

        // GlobalModel: eg = seg_mean(e, batch[row]); u' = mlp(concat(u, eg))
        zero_f32<<<gsz((long)NG * s.eo), 256, 0, stream>>>(EG, (long)NG * s.eo);
        zero_f32<<<gsz(NG), 256, 0, stream>>>(GCNT, NG);
        seg_scatter<<<gsz((long)NE * s.eo), 256, 0, stream>>>(Eout, rowIdx, batch, EG, GCNT, NE, s.eo);
        seg_div<<<gsz((long)NG * s.eo), 256, 0, stream>>>(EG, GCNT, NG, s.eo);
        glob_mlp<<<NG, 128, 0, stream>>>(Uin, EG, s.gf, s.eo, gw1, gb1, h1, gw2, gb2, s.gf, Uout);
    };

    auto run_bn = [&](float* H, long M, int F, const float* g, const float* b) {
        zero_f32<<<gsz(2 * F), 256, 0, stream>>>(STAT, 2 * F);
        bn_stats<<<512, 256, 0, stream>>>(H, M, F, STAT);
        bn_apply<<<gsz(M * F), 256, 0, stream>>>(H, M, F, STAT, g, b);
    };

    // layer 0
    run_layer(0, x, edge_attr, U0, XB0, EB0, U1, false);
    run_bn(XB0, NN, 256, nbn1_g, nbn1_b);
    run_bn(EB0, NE, 128, ebn1_g, ebn1_b);
    // layer 1
    run_layer(1, XB0, EB0, U1, XB1, EB1, U0, false);
    run_bn(XB1, NN, 512, nbn2_g, nbn2_b);
    run_bn(EB1, NE, 256, ebn2_g, ebn2_b);
    // layer 2 (residuals)
    run_layer(2, XB1, EB1, U0, XB0, EB0, U1, true);

    // head
    fc_head<<<NG, 32, 0, stream>>>(U1, fc1_w, fc1_b, fc2_w, fc2_b, (float*)d_out);
}
